// Model_34402688041530
// MI455X (gfx1250) — matrix operand order repaired and checked by reading, not compiled, not run on hardware
//
#include <hip/hip_runtime.h>
#include <hip/hip_bf16.h>
#include <math.h>

// ---------------------------------------------------------------------------
// Seq2seq RNN+attention forward loss for MI455X (gfx1250, wave32, WMMA+TDM).
//   - All batched matmuls: v_wmma_f32_16x16x32_bf16, bf16-resident operands.
//   - Logits GEMM (512x32000x1024) stages its B panel into LDS with the
//     Tensor Data Mover (tensor_load_to_lds + s_wait_tensorcnt) and keeps a
//     streaming logsumexp tile epilogue (never materializes 65MB of logits).
//   - Sequential RNN scans: persistent cooperative kernels + grid barrier.
// ---------------------------------------------------------------------------

typedef __attribute__((ext_vector_type(16))) __bf16 v16bf;
typedef __attribute__((ext_vector_type(8)))  __bf16 v8bf;
typedef __attribute__((ext_vector_type(8)))  float  v8f;
typedef unsigned int u32;
typedef __attribute__((ext_vector_type(4)))  u32 u32x4;
typedef __attribute__((ext_vector_type(8)))  int i32x8;
typedef __attribute__((ext_vector_type(4)))  int i32x4;

#define HDIM   1024
#define SLEN   512
#define TLEN   512
#define VOC    32000
#define NTILEV (VOC / 16)      // 2000 column tiles of the vocab
#define LCOL   2               // 16-col tiles per wave in the logits kernel
#define SCAN_BLOCKS 64

// ---------------- WMMA fragment loaders (bf16-resident operands) ------------
// A-matrix 16x32 bf16: lane L holds row m=L%16;
//   lanes 0-15: K in {0..15}; lanes 16-31: {16..31} (matches B).
__device__ __forceinline__ v16bf load_fragA_bf(const __bf16* __restrict__ rowp,
                                               int kt, int hi) {
  const v8bf* p0 = (const v8bf*)(rowp + kt + (hi << 4));
  const v8bf* p1 = (const v8bf*)(rowp + kt + (hi << 4) + 8);
  v8bf a0 = *p0, a1 = *p1;
  return __builtin_shufflevector(a0, a1, 0, 1, 2, 3, 4, 5, 6, 7,
                                 8, 9, 10, 11, 12, 13, 14, 15);
}
// B-matrix 32x16 bf16 (K x N): lane L holds column n=L%16; lanes 0-15 hold
// K=0..15, lanes 16-31 hold K=16..31 (one contiguous 16-element run).
// We compute D = A * W^T with W row-major [N,K] -> lane reads W[n][kt+hi*16..].
__device__ __forceinline__ v16bf load_fragB_bf(const __bf16* __restrict__ rowp,
                                               int kt, int hi) {
  const v8bf* p0 = (const v8bf*)(rowp + kt + (hi << 4));
  v8bf b0 = p0[0], b1 = p0[1];
  return __builtin_shufflevector(b0, b1, 0, 1, 2, 3, 4, 5, 6, 7,
                                 8, 9, 10, 11, 12, 13, 14, 15);
}

// ---------------- Tensor Data Mover: stage nrows x K bf16 rows into LDS -----
// D# per CDNA5 ISA ch.8: group0 {count=1, lds_addr, global_addr, type=2},
// group1 {data_size=2B, tensor_dim0=K, tensor_dim1=VOC, tile_dim0=K,
//         tile_dim1=nrows, tensor_dim0_stride=K}; groups 2/3 zero (2-D).
__device__ __forceinline__ void tdm_load_rows_to_lds(
    u32 lds_off, const __bf16* gbase, int rowStart, int nrows, int K) {
  unsigned long long ga = (unsigned long long)(size_t)gbase +
                          (unsigned long long)rowStart * (unsigned)K * 2ull;
  u32x4 g0;
  g0[0] = 1u;                                        // count=1, user mode
  g0[1] = lds_off;                                   // LDS byte address
  g0[2] = (u32)ga;                                   // global_addr[31:0]
  g0[3] = (u32)((ga >> 32) & 0x01FFFFFFu) | 0x80000000u;  // [56:32] | type=2
  u32 td0 = (u32)K, td1 = (u32)VOC, tl0 = (u32)K, tl1 = (u32)nrows;
  i32x8 g1;
  g1[0] = (int)(1u << 16);                           // data_size = 2 bytes
  g1[1] = (int)((td0 & 0xFFFFu) << 16);              // tensor_dim0 lo
  g1[2] = (int)((td0 >> 16) | ((td1 & 0xFFFFu) << 16));
  g1[3] = (int)((td1 >> 16) | (tl0 << 16));          // tile_dim0
  g1[4] = (int)tl1;                                  // tile_dim1 (tile_dim2=0)
  g1[5] = (int)td0;                                  // tensor_dim0_stride lo48
  g1[6] = 0;
  g1[7] = 0;
  i32x4 z4 = (i32x4)0;
  i32x8 z8 = (i32x8)0;
  // 6-arg form (clang-23 / therock headers): (g0, g1, g2, g3, g4, cpol)
  __builtin_amdgcn_tensor_load_to_lds(g0, g1, z4, z4, z8, 0);
}

// ---------------- generic GEMM: out[M,N] = op(A[M,K] * W[N,K]^T + bias) -----
__global__ __launch_bounds__(32) void gemm_bf16_kernel(
    const __bf16* __restrict__ A, const __bf16* __restrict__ W,
    const float* __restrict__ bias, float* __restrict__ out,
    int M, int N, int K, int do_tanh) {
  int ntiles = N >> 4;
  int tn = (blockIdx.x % ntiles) << 4;
  int tm = (blockIdx.x / ntiles) << 4;
  int lane = threadIdx.x & 31;
  int lo = lane & 15, hi = lane >> 4;
  const __bf16* arow = A + (size_t)(tm + lo) * K;
  const __bf16* wrow = W + (size_t)(tn + lo) * K;
  v8f acc = {};
  for (int kt = 0; kt < K; kt += 32) {
    __builtin_prefetch(wrow + kt + 256, 0, 1);       // global_prefetch_b8
    v16bf a = load_fragA_bf(arow, kt, hi);
    v16bf b = load_fragB_bf(wrow, kt, hi);
    acc = __builtin_amdgcn_wmma_f32_16x16x32_bf16(false, a, false, b,
                                                  (short)0, acc, false, false);
  }
  float bv = bias ? bias[tn + lo] : 0.0f;
  for (int d = 0; d < 8; ++d) {
    int m = tm + d + (hi << 3);        // D layout: VGPR d -> row d (+8 hi half)
    float v = acc[d] + bv;
    if (do_tanh) v = tanhf(v);
    out[(size_t)m * N + tn + lo] = v;
  }
}

// ------- logits GEMM: TDM-staged B panel + per-tile (rowmax, sumexp) --------
__global__ __launch_bounds__(32) void logits_lse_tdm_kernel(
    const __bf16* __restrict__ Zb, const __bf16* __restrict__ Wb,
    const float* __restrict__ bout, float* __restrict__ pMax,
    float* __restrict__ pSum, int M, int N, int K) {
  __shared__ __bf16 ldsB[16 * LCOL * HDIM];          // 32 rows x 1024 = 64 KB
  int ntiles = N >> 4;                               // 2000
  int ngrp = ntiles / LCOL;                          // 1000
  int grp = blockIdx.x % ngrp;
  int tm  = (blockIdx.x / ngrp) << 4;
  int rowStart = grp * (16 * LCOL);                  // first staged vocab row
  int lane = threadIdx.x & 31;
  int lo = lane & 15, hi = lane >> 4;

  // Tensor DMA: 32 vocab rows x K bf16 -> LDS (issued once per wave)
  tdm_load_rows_to_lds((u32)(size_t)ldsB, Wb, rowStart, 16 * LCOL, K);

  const __bf16* arow = Zb + (size_t)(tm + lo) * K;
  v8f acc[LCOL];
  #pragma unroll
  for (int c = 0; c < LCOL; ++c) acc[c] = (v8f)0.0f;

  __builtin_amdgcn_s_wait_tensorcnt(0);
  for (int kt = 0; kt < K; kt += 32) {
    v16bf a = load_fragA_bf(arow, kt, hi);
    #pragma unroll
    for (int c = 0; c < LCOL; ++c) {
      const v8bf* bp = (const v8bf*)&ldsB[(size_t)(c * 16 + lo) * K + kt + (hi << 4)];
      v8bf b0 = bp[0], b1 = bp[1];
      v16bf b = __builtin_shufflevector(b0, b1, 0, 1, 2, 3, 4, 5, 6, 7,
                                        8, 9, 10, 11, 12, 13, 14, 15);
      acc[c] = __builtin_amdgcn_wmma_f32_16x16x32_bf16(
          false, a, false, b, (short)0, acc[c], false, false);
    }
  }
  #pragma unroll
  for (int c = 0; c < LCOL; ++c) {
    int tcol = grp * LCOL + c;
    float bv = bout[(tcol << 4) + lo];
    for (int d = 0; d < 8; ++d) {
      float v = acc[c][d] + bv;
      float mx = v;                     // row-wise reduce across 16 lanes
      for (int s = 1; s < 16; s <<= 1) mx = fmaxf(mx, __shfl_xor(mx, s, 16));
      float e = __expf(v - mx);
      for (int s = 1; s < 16; s <<= 1) e += __shfl_xor(e, s, 16);
      if (lo == 0) {
        int m = tm + d + (hi << 3);
        pMax[(size_t)m * ntiles + tcol] = mx;
        pSum[(size_t)m * ntiles + tcol] = e;
      }
    }
  }
}

// ---------------- cooperative grid barrier ----------------------------------
__device__ __forceinline__ void gridBarrier(unsigned* cnt, unsigned* gen,
                                            unsigned nblk, unsigned step) {
  __syncthreads();
  if (threadIdx.x == 0) {
    __threadfence();
    unsigned prev = atomicAdd(cnt, 1u);
    if (prev == nblk * (step + 1u) - 1u) atomicAdd(gen, 1u);
    while (atomicAdd(gen, 0u) < step + 1u) __builtin_amdgcn_s_sleep(1);
    __threadfence();
  }
  __syncthreads();
}

// ------- sequential tanh-RNN scan: h_t = tanh(xW[t] + Wh * h_{t-1}) ---------
__global__ __launch_bounds__(256) void rnn_scan_kernel(
    const float* __restrict__ Wh, const float* __restrict__ xW,
    float* __restrict__ hbuf /*[2*HDIM], [0] preloaded*/,
    float* __restrict__ Hout, int T, int writePrev,
    unsigned* __restrict__ barCnt, unsigned* __restrict__ barGen) {
  __shared__ float lh[HDIM + 16];            // padded: idx + idx/64
  __shared__ float part[16][17];
  int tid = threadIdx.x;
  int jj = tid >> 4, kk = tid & 15;
  int gj = (blockIdx.x << 4) + jj;           // this block owns 16 outputs
  const float* wrow = Wh + (size_t)gj * HDIM + (kk << 6);
  for (int t = 0; t < T; ++t) {
    const float* hin = hbuf + ((t & 1) ? HDIM : 0);
    for (int i = tid; i < HDIM; i += 256) {
      float v = hin[i];
      lh[i + (i >> 6)] = v;
      if (writePrev && blockIdx.x == 0) Hout[(size_t)t * HDIM + i] = v;
    }
    __syncthreads();
    float sum = 0.0f;
    int lb = kk * 65;                        // padded base of this K-chunk
    #pragma unroll 8
    for (int i = 0; i < 64; ++i) sum += wrow[i] * lh[lb + i];
    part[jj][kk] = sum;
    __syncthreads();
    if (kk == 0) {
      float s2 = 0.0f;
      #pragma unroll
      for (int q = 0; q < 16; ++q) s2 += part[jj][q];
      float hnew = tanhf(xW[(size_t)t * HDIM + gj] + s2);
      hbuf[(((t + 1) & 1) ? HDIM : 0) + gj] = hnew;
      if (!writePrev) Hout[(size_t)t * HDIM + gj] = hnew;
    }
    gridBarrier(barCnt, barGen, SCAN_BLOCKS, (unsigned)t);
  }
}

// ---------------- small helper kernels --------------------------------------
__global__ void init_kernel(unsigned* counters) {
  if (threadIdx.x < 8) counters[threadIdx.x] = 0u;
}
__global__ void copy_vec_kernel(const float* __restrict__ src,
                                float* __restrict__ dst, int n) {
  int i = blockIdx.x * blockDim.x + threadIdx.x;
  if (i < n) dst[i] = src[i];
}
__global__ void f32_to_bf16_kernel(const float* __restrict__ in,
                                   __bf16* __restrict__ out, int n) {
  int i = blockIdx.x * blockDim.x + threadIdx.x;
  if (i < n) out[i] = (__bf16)in[i];
}
__global__ void gather_rows_bf16_kernel(const float* __restrict__ table,
                                        const int* __restrict__ idx,
                                        __bf16* __restrict__ out, int total,
                                        int h) {
  int i = blockIdx.x * blockDim.x + threadIdx.x;
  if (i < total) {
    int r = i / h, c = i - r * h;
    out[i] = (__bf16)table[(size_t)idx[r] * h + c];
  }
}
__global__ void concat_bf16_kernel(const float* __restrict__ C,
                                   const float* __restrict__ Hp,
                                   __bf16* __restrict__ Cat, int T, int h) {
  int i = blockIdx.x * blockDim.x + threadIdx.x;
  if (i < T * 2 * h) {
    int t = i / (2 * h), j = i - t * 2 * h;
    float v = (j < h) ? C[(size_t)t * h + j] : Hp[(size_t)t * h + (j - h)];
    Cat[i] = (__bf16)v;
  }
}
__global__ __launch_bounds__(256) void softmax512_kernel(
    const float* __restrict__ S, __bf16* __restrict__ Wsm) {
  __shared__ float red[256];
  int t = blockIdx.x, tid = threadIdx.x;
  const float* row = S + (size_t)t * 512;
  float a = row[tid], b = row[tid + 256];
  red[tid] = fmaxf(a, b);
  __syncthreads();
  for (int s = 128; s > 0; s >>= 1) {
    if (tid < s) red[tid] = fmaxf(red[tid], red[tid + s]);
    __syncthreads();
  }
  float mx = red[0];
  __syncthreads();
  float ea = __expf(a - mx), eb = __expf(b - mx);
  red[tid] = ea + eb;
  __syncthreads();
  for (int s = 128; s > 0; s >>= 1) {
    if (tid < s) red[tid] += red[tid + s];
    __syncthreads();
  }
  float inv = 1.0f / red[0];
  Wsm[(size_t)t * 512 + tid] = (__bf16)(ea * inv);
  Wsm[(size_t)t * 512 + tid + 256] = (__bf16)(eb * inv);
}
__global__ __launch_bounds__(256) void transpose_bf16_kernel(
    const float* __restrict__ in, __bf16* __restrict__ out, int R, int Cc) {
  __shared__ float tile[32][33];
  int nbx = Cc / 32;
  int c0 = (blockIdx.x % nbx) * 32, r0 = (blockIdx.x / nbx) * 32;
  int tx = threadIdx.x & 31, ty = threadIdx.x >> 5;
  for (int i = 0; i < 32; i += 8)
    tile[ty + i][tx] = in[(size_t)(r0 + ty + i) * Cc + c0 + tx];
  __syncthreads();
  for (int i = 0; i < 32; i += 8)
    out[(size_t)(c0 + ty + i) * R + r0 + tx] = (__bf16)tile[tx][ty + i];
}
__global__ __launch_bounds__(256) void loss_kernel(
    const float* __restrict__ pMax, const float* __restrict__ pSum,
    const float* __restrict__ Z, const float* __restrict__ Wout,
    const float* __restrict__ bout, const int* __restrict__ tgt,
    float* __restrict__ losses, int NT, int K) {
  __shared__ float red[256];
  int t = blockIdx.x, tid = threadIdx.x;
  const float* pm = pMax + (size_t)t * NT;
  const float* ps = pSum + (size_t)t * NT;
  float mx = -INFINITY;
  for (int i = tid; i < NT; i += 256) mx = fmaxf(mx, pm[i]);
  red[tid] = mx;
  __syncthreads();
  for (int s = 128; s > 0; s >>= 1) {
    if (tid < s) red[tid] = fmaxf(red[tid], red[tid + s]);
    __syncthreads();
  }
  mx = red[0];
  __syncthreads();
  float sum = 0.0f;
  for (int i = tid; i < NT; i += 256) sum += ps[i] * __expf(pm[i] - mx);
  red[tid] = sum;
  __syncthreads();
  for (int s = 128; s > 0; s >>= 1) {
    if (tid < s) red[tid] += red[tid + s];
    __syncthreads();
  }
  float lse = mx + __logf(red[0]);
  __syncthreads();
  int v = tgt[t];
  const float* zr = Z + (size_t)t * K;
  const float* wr = Wout + (size_t)v * K;
  float dot = 0.0f;
  for (int i = tid; i < K; i += 256) dot += zr[i] * wr[i];
  red[tid] = dot;
  __syncthreads();
  for (int s = 128; s > 0; s >>= 1) {
    if (tid < s) red[tid] += red[tid + s];
    __syncthreads();
  }
  if (tid == 0) losses[t] = lse - (red[0] + bout[v]);
}
__global__ __launch_bounds__(256) void sum_kernel(
    const float* __restrict__ losses, float* __restrict__ out, int n) {
  __shared__ float red[256];
  int tid = threadIdx.x;
  float s = 0.0f;
  for (int i = tid; i < n; i += 256) s += losses[i];
  red[tid] = s;
  __syncthreads();
  for (int k = 128; k > 0; k >>= 1) {
    if (tid < k) red[tid] += red[tid + k];
    __syncthreads();
  }
  if (tid == 0) out[0] = red[0];
}

// ---------------------------------------------------------------------------
extern "C" void kernel_launch(void* const* d_in, const int* in_sizes, int n_in,
                              void* d_out, int out_size, void* d_ws,
                              size_t ws_size, hipStream_t stream) {
  const int*   src_nums = (const int*)d_in[0];
  const int*   tgt_nums = (const int*)d_in[1];
  const float* src_emb  = (const float*)d_in[2];
  const float* enc_Wx1  = (const float*)d_in[3];
  const float* enc_Wh1  = (const float*)d_in[4];
  const float* enc_b1   = (const float*)d_in[5];
  const float* enc_h01  = (const float*)d_in[6];
  const float* enc_Wx2  = (const float*)d_in[7];
  const float* enc_Wh2  = (const float*)d_in[8];
  const float* enc_b2   = (const float*)d_in[9];
  const float* enc_h02  = (const float*)d_in[10];
  const float* tgt_emb  = (const float*)d_in[11];
  const float* dec_h0   = (const float*)d_in[12];
  const float* dec_Wx   = (const float*)d_in[13];
  const float* dec_Wh   = (const float*)d_in[14];
  const float* dec_b    = (const float*)d_in[15];
  const float* tnh_W    = (const float*)d_in[16];
  const float* tnh_b    = (const float*)d_in[17];
  const float* out_W    = (const float*)d_in[18];
  const float* out_b    = (const float*)d_in[19];

  char* w = (char*)d_ws;
  size_t off = 0;
  auto alloc = [&](size_t bytes) -> void* {
    void* p = w + off;
    off = (off + bytes + 255) & ~(size_t)255;
    return p;
  };
  // fp32 intermediates
  float* xW1   = (float*)alloc((size_t)SLEN * HDIM * 4);
  float* H1    = (float*)alloc((size_t)SLEN * HDIM * 4);
  float* xW2   = (float*)alloc((size_t)SLEN * HDIM * 4);
  float* Henc  = (float*)alloc((size_t)SLEN * HDIM * 4);
  float* xWd   = (float*)alloc((size_t)TLEN * HDIM * 4);
  float* Hprev = (float*)alloc((size_t)TLEN * HDIM * 4);
  float* Smat  = (float*)alloc((size_t)TLEN * SLEN * 4);
  float* Cmat  = (float*)alloc((size_t)TLEN * HDIM * 4);
  float* Zmat  = (float*)alloc((size_t)TLEN * HDIM * 4);
  float* pMax  = (float*)alloc((size_t)TLEN * NTILEV * 4);
  float* pSum  = (float*)alloc((size_t)TLEN * NTILEV * 4);
  float* losses = (float*)alloc((size_t)TLEN * 4);
  float* hbuf   = (float*)alloc((size_t)2 * HDIM * 4);
  unsigned* counters = (unsigned*)alloc(8 * 4);   // 3 pairs (cnt, gen)
  // bf16 operands
  __bf16* embSb  = (__bf16*)alloc((size_t)SLEN * HDIM * 2);
  __bf16* embTb  = (__bf16*)alloc((size_t)TLEN * HDIM * 2);
  __bf16* H1b    = (__bf16*)alloc((size_t)SLEN * HDIM * 2);
  __bf16* Hencb  = (__bf16*)alloc((size_t)SLEN * HDIM * 2);
  __bf16* HencTb = (__bf16*)alloc((size_t)HDIM * SLEN * 2);
  __bf16* Hprevb = (__bf16*)alloc((size_t)TLEN * HDIM * 2);
  __bf16* Wsmb   = (__bf16*)alloc((size_t)TLEN * SLEN * 2);
  __bf16* Catb   = (__bf16*)alloc((size_t)TLEN * 2 * HDIM * 2);
  __bf16* Zb     = (__bf16*)alloc((size_t)TLEN * HDIM * 2);
  __bf16* Wx1b   = (__bf16*)alloc((size_t)HDIM * HDIM * 2);
  __bf16* Wx2b   = (__bf16*)alloc((size_t)HDIM * HDIM * 2);
  __bf16* dWxb   = (__bf16*)alloc((size_t)HDIM * HDIM * 2);
  __bf16* tnhWb  = (__bf16*)alloc((size_t)HDIM * 2 * HDIM * 2);
  __bf16* outWb  = (__bf16*)alloc((size_t)VOC * HDIM * 2);

  const int HH = HDIM * HDIM;
  init_kernel<<<1, 64, 0, stream>>>(counters);

  // weight conversions (reused operands -> bf16 once)
  f32_to_bf16_kernel<<<HH / 256, 256, 0, stream>>>(enc_Wx1, Wx1b, HH);
  f32_to_bf16_kernel<<<HH / 256, 256, 0, stream>>>(enc_Wx2, Wx2b, HH);
  f32_to_bf16_kernel<<<HH / 256, 256, 0, stream>>>(dec_Wx, dWxb, HH);
  f32_to_bf16_kernel<<<2 * HH / 256, 256, 0, stream>>>(tnh_W, tnhWb, 2 * HH);
  f32_to_bf16_kernel<<<(VOC * HDIM) / 256, 256, 0, stream>>>(out_W, outWb,
                                                             VOC * HDIM);

  // ---------------- encoder ----------------
  gather_rows_bf16_kernel<<<(SLEN * HDIM) / 256, 256, 0, stream>>>(
      src_emb, src_nums, embSb, SLEN * HDIM, HDIM);
  gemm_bf16_kernel<<<(SLEN / 16) * (HDIM / 16), 32, 0, stream>>>(
      embSb, Wx1b, enc_b1, xW1, SLEN, HDIM, HDIM, 0);
  copy_vec_kernel<<<4, 256, 0, stream>>>(enc_h01, hbuf, HDIM);
  rnn_scan_kernel<<<SCAN_BLOCKS, 256, 0, stream>>>(
      enc_Wh1, xW1, hbuf, H1, SLEN, 0, counters + 0, counters + 1);
  f32_to_bf16_kernel<<<(SLEN * HDIM) / 256, 256, 0, stream>>>(H1, H1b,
                                                              SLEN * HDIM);
  gemm_bf16_kernel<<<(SLEN / 16) * (HDIM / 16), 32, 0, stream>>>(
      H1b, Wx2b, enc_b2, xW2, SLEN, HDIM, HDIM, 0);
  copy_vec_kernel<<<4, 256, 0, stream>>>(enc_h02, hbuf, HDIM);
  rnn_scan_kernel<<<SCAN_BLOCKS, 256, 0, stream>>>(
      enc_Wh2, xW2, hbuf, Henc, SLEN, 0, counters + 2, counters + 3);

  // ---------------- decoder state scan (teacher-forced, independent) -------
  gather_rows_bf16_kernel<<<(TLEN * HDIM) / 256, 256, 0, stream>>>(
      tgt_emb, tgt_nums, embTb, TLEN * HDIM, HDIM);
  gemm_bf16_kernel<<<(TLEN / 16) * (HDIM / 16), 32, 0, stream>>>(
      embTb, dWxb, dec_b, xWd, TLEN, HDIM, HDIM, 0);
  copy_vec_kernel<<<4, 256, 0, stream>>>(dec_h0, hbuf, HDIM);
  rnn_scan_kernel<<<SCAN_BLOCKS, 256, 0, stream>>>(
      dec_Wh, xWd, hbuf, Hprev, TLEN, 1, counters + 4, counters + 5);

  // ---------------- batched attention + output head ------------------------
  f32_to_bf16_kernel<<<(SLEN * HDIM) / 256, 256, 0, stream>>>(Henc, Hencb,
                                                              SLEN * HDIM);
  f32_to_bf16_kernel<<<(TLEN * HDIM) / 256, 256, 0, stream>>>(Hprev, Hprevb,
                                                              TLEN * HDIM);
  gemm_bf16_kernel<<<(TLEN / 16) * (SLEN / 16), 32, 0, stream>>>(
      Hprevb, Hencb, (const float*)nullptr, Smat, TLEN, SLEN, HDIM, 0);
  softmax512_kernel<<<TLEN, 256, 0, stream>>>(Smat, Wsmb);
  transpose_bf16_kernel<<<(HDIM / 32) * (SLEN / 32), 256, 0, stream>>>(
      Henc, HencTb, SLEN, HDIM);
  gemm_bf16_kernel<<<(TLEN / 16) * (HDIM / 16), 32, 0, stream>>>(
      Wsmb, HencTb, (const float*)nullptr, Cmat, TLEN, HDIM, SLEN, 0);
  concat_bf16_kernel<<<(TLEN * 2 * HDIM) / 256, 256, 0, stream>>>(
      Cmat, Hprev, Catb, TLEN, HDIM);
  gemm_bf16_kernel<<<(TLEN / 16) * (HDIM / 16), 32, 0, stream>>>(
      Catb, tnhWb, tnh_b, Zmat, TLEN, HDIM, 2 * HDIM, 1);
  f32_to_bf16_kernel<<<(TLEN * HDIM) / 256, 256, 0, stream>>>(Zmat, Zb,
                                                              TLEN * HDIM);

  // big GEMM 512 x 32000 x 1024: TDM-staged B + streaming logsumexp partials
  logits_lse_tdm_kernel<<<(TLEN / 16) * (NTILEV / LCOL), 32, 0, stream>>>(
      Zb, outWb, out_b, pMax, pSum, TLEN, VOC, HDIM);
  loss_kernel<<<TLEN, 256, 0, stream>>>(pMax, pSum, Zmat, out_W, out_b,
                                        tgt_nums, losses, NTILEV, HDIM);
  sum_kernel<<<1, 256, 0, stream>>>(losses, (float*)d_out, TLEN);
}
